// ViM_MambaBlock_11252814315696
// MI455X (gfx1250) — matrix operand order repaired and checked by reading, not compiled, not run on hardware
//
#include <hip/hip_runtime.h>
#include <hip/hip_bf16.h>

// ---------------------------------------------------------------------------
// ViM Mamba block for MI455X (gfx1250, wave32, WMMA + async LDS staging).
// Dims fixed by the reference: D=1024, DI=2048, N=16, R=64, K=4, B=2, S=2048.
// ---------------------------------------------------------------------------

#define D_MODEL 1024
#define D_INNER 2048
#define N_STATE 16
#define DT_RANK 64
#define K_CONV 4
#define BATCH 2
#define SEQ 2048
#define ROWS (BATCH * SEQ)          // 4096 token rows
#define XP_PAD 128                  // x_proj out cols padded 96 -> 128

typedef _Float16 half_t;
typedef __attribute__((ext_vector_type(16))) _Float16 v16h;
typedef __attribute__((ext_vector_type(8)))  float    v8f;
typedef __attribute__((ext_vector_type(4)))  int      v4i_t;

#define AS_GLOBAL __attribute__((address_space(1)))
#define AS_LDS    __attribute__((address_space(3)))

#if __has_builtin(__builtin_amdgcn_global_load_async_to_lds_b128)
#define HAVE_ASYNC_LDS 1
#endif

union AFragU { uint4 u4[2]; v16h h; };

__device__ __forceinline__ float sigmoidf_(float x) {
    return 1.0f / (1.0f + __expf(-x));
}
__device__ __forceinline__ float siluf_(float x) {
    return x * sigmoidf_(x);
}
__device__ __forceinline__ float softplusf_(float x) {
    return (x > 20.0f) ? x : log1pf(__expf(x));
}

__device__ __forceinline__ void wait_async_le1(void) {
#if defined(HAVE_ASYNC_LDS)
#if __has_builtin(__builtin_amdgcn_s_wait_asynccnt)
    __builtin_amdgcn_s_wait_asynccnt(1);
#else
    asm volatile("s_wait_asynccnt 0x1" ::: "memory");
#endif
#endif
}
__device__ __forceinline__ void wait_async_le0(void) {
#if defined(HAVE_ASYNC_LDS)
#if __has_builtin(__builtin_amdgcn_s_wait_asynccnt)
    __builtin_amdgcn_s_wait_asynccnt(0);
#else
    asm volatile("s_wait_asynccnt 0x0" ::: "memory");
#endif
#endif
}

#if defined(HAVE_ASYNC_LDS)
__device__ __forceinline__ void async_copy_b128(const half_t* gsrc, half_t* ldst) {
    __builtin_amdgcn_global_load_async_to_lds_b128(
        (AS_GLOBAL v4i_t*)(AS_GLOBAL void*)(gsrc),
        (AS_LDS    v4i_t*)(AS_LDS    void*)(ldst), 0, 0);
}
#else
__device__ __forceinline__ void async_copy_b128(const half_t* gsrc, half_t* ldst) {
    *(uint4*)ldst = *(const uint4*)gsrc;
}
#endif

// ---------------------------------------------------------------------------
// 1) adaLN modulation: mod[b,e] = sum_c silu(c[b,c]) * ada_w[e,c] + ada_b[e]
// ---------------------------------------------------------------------------
__global__ void k_adaln(const float* __restrict__ c,
                        const float* __restrict__ ada_w,
                        const float* __restrict__ ada_b,
                        float* __restrict__ mod) {
    __shared__ float sc[2 * D_MODEL];
    const int b = blockIdx.y;
    const int e = blockIdx.x * blockDim.x + threadIdx.x;   // 0..3071
    for (int i = threadIdx.x; i < 2 * D_MODEL; i += blockDim.x)
        sc[i] = siluf_(c[b * 2 * D_MODEL + i]);
    __syncthreads();
    const float* wr = ada_w + (size_t)e * (2 * D_MODEL);
    float acc = 0.0f;
    #pragma unroll 8
    for (int i = 0; i < 2 * D_MODEL; ++i) acc = fmaf(sc[i], wr[i], acc);
    mod[b * 3 * D_MODEL + e] = acc + ada_b[e];
}

// ---------------------------------------------------------------------------
// 2) LayerNorm + modulate -> f16 activations  xs_h:[ROWS, D]
// ---------------------------------------------------------------------------
__global__ void k_ln_mod(const float* __restrict__ x,
                         const float* __restrict__ ln_w,
                         const float* __restrict__ ln_b,
                         const float* __restrict__ mod,
                         half_t* __restrict__ xs_h) {
    __shared__ float red[256];
    const int row = blockIdx.x;            // 0..ROWS-1
    const int b = row / SEQ;
    const int t = threadIdx.x;
    const float* xr = x + (size_t)row * D_MODEL;

    float v[4];
    float s = 0.0f;
    #pragma unroll
    for (int i = 0; i < 4; ++i) { v[i] = xr[t + i * 256]; s += v[i]; }
    red[t] = s; __syncthreads();
    for (int off = 128; off > 0; off >>= 1) {
        if (t < off) red[t] += red[t + off];
        __syncthreads();
    }
    const float mu = red[0] * (1.0f / D_MODEL);
    __syncthreads();
    float s2 = 0.0f;
    #pragma unroll
    for (int i = 0; i < 4; ++i) { float d = v[i] - mu; s2 += d * d; }
    red[t] = s2; __syncthreads();
    for (int off = 128; off > 0; off >>= 1) {
        if (t < off) red[t] += red[t + off];
        __syncthreads();
    }
    const float rstd = rsqrtf(red[0] * (1.0f / D_MODEL) + 1e-5f);
    const float* mrow = mod + (size_t)b * 3 * D_MODEL;
    #pragma unroll
    for (int i = 0; i < 4; ++i) {
        const int d = t + i * 256;
        float xn = (v[i] - mu) * rstd * ln_w[d] + ln_b[d];
        float xs = xn * (1.0f + mrow[D_MODEL + d]) + mrow[d];
        xs_h[(size_t)row * D_MODEL + d] = (half_t)xs;
    }
}

// ---------------------------------------------------------------------------
// 3) weight repacks f32 -> f16
// ---------------------------------------------------------------------------
__global__ void k_cvt16(const float* __restrict__ src, half_t* __restrict__ dst, int n) {
    int i = blockIdx.x * blockDim.x + threadIdx.x;
    if (i < n) dst[i] = (half_t)src[i];
}
// x_proj_w [96, 2048] -> zero-padded [128, 2048]
__global__ void k_cvt16_padrows(const float* __restrict__ src, half_t* __restrict__ dst) {
    int i = blockIdx.x * blockDim.x + threadIdx.x;     // over 128*2048
    int r = i / D_INNER, col = i % D_INNER;
    dst[i] = (r < (DT_RANK + 2 * N_STATE)) ? (half_t)src[r * D_INNER + col] : (half_t)0.0f;
}

// ---------------------------------------------------------------------------
// 4) WMMA GEMM:  C[m,n] = sum_k A[m,k] * W[n,k]     (W row-major [Ncols,K])
//    Block = 256 threads (8 waves) -> 128(M) x 64(N) tile.
//    The 64x32 f16 weight slice per K-step is staged once per block into LDS
//    (double-buffered, GLOBAL_LOAD_ASYNC_TO_LDS_B128, one 16B chunk/thread),
//    shared by all 8 waves. A-fragments stream from global with prefetch.
//    K % 32 == 0, Ncols % 64 == 0, M % 128 == 0.
// ---------------------------------------------------------------------------
__global__ void k_wmma_gemm(const half_t* __restrict__ A,
                            const half_t* __restrict__ W,
                            float* __restrict__ C,
                            int K, int ldC, int tilesN) {
    __shared__ __align__(32) half_t lds_w[2][64 * 32];   // [buf][n*32 + k], 8 KB

    const int tid  = threadIdx.x;
    const int wave = tid >> 5;
    const int lane = tid & 31;
    const int half = lane >> 4;          // 0 | 1
    const int l16  = lane & 15;

    const int bm = blockIdx.x / tilesN;
    const int bn = blockIdx.x % tilesN;
    const int n0 = bn * 64;
    const int m0 = bm * 128 + wave * 16;

    // Per-thread staging chunk: 256 chunks of 16B cover 64 rows x 32 k f16.
    const int n_s  = tid >> 2;               // 0..63
    const int ko_s = (tid & 3) * 8;          // f16 elements within the 32-k slice
    const half_t* wsrc = W + (size_t)(n0 + n_s) * K + ko_s;
    half_t* ldst0 = &lds_w[0][n_s * 32 + ko_s];
    half_t* ldst1 = &lds_w[1][n_s * 32 + ko_s];

    // A row for this lane (ISA 7.12.2 A-fragment layout).
    const half_t* arow = A + (size_t)(m0 + l16) * K;

    v8f acc0 = {}, acc1 = {}, acc2 = {}, acc3 = {};

    const int nsteps = K >> 5;

    // Prologue: stage k-slice 0 into buffer 0.
    async_copy_b128(wsrc, ldst0);

    for (int s = 0; s < nsteps; ++s) {
        const int k0 = s << 5;
        const bool has_next = (s + 1 < nsteps);
        half_t* ldst_next = ((s + 1) & 1) ? ldst1 : ldst0;

        if (has_next) {
            async_copy_b128(wsrc + k0 + 32, ldst_next);
            wait_async_le1();
        } else {
            wait_async_le0();
        }
        __syncthreads();   // current buffer fully staged for all waves

        // A fragment: V0..7 = K[k0 + half*16 .. +15]
        AFragU a;
        const uint4* ap = reinterpret_cast<const uint4*>(arow + k0 + half * 16);
        a.u4[0] = ap[0];
        a.u4[1] = ap[1];   // +8 f16 elements
        if (has_next) __builtin_prefetch(arow + k0 + 32, 0, 1);

        // B fragments from LDS: lane holds W[n0+j*16+l16, k0 + half*16 .. +15]
        const half_t* lw = &lds_w[s & 1][0];
        const v16h b0 = *(const v16h*)(lw + (l16)      * 32 + half * 16);
        const v16h b1 = *(const v16h*)(lw + (16 + l16) * 32 + half * 16);
        const v16h b2 = *(const v16h*)(lw + (32 + l16) * 32 + half * 16);
        const v16h b3 = *(const v16h*)(lw + (48 + l16) * 32 + half * 16);

        acc0 = __builtin_amdgcn_wmma_f32_16x16x32_f16(false, a.h, false, b0, (short)0, acc0, false, false);
        acc1 = __builtin_amdgcn_wmma_f32_16x16x32_f16(false, a.h, false, b1, (short)0, acc1, false, false);
        acc2 = __builtin_amdgcn_wmma_f32_16x16x32_f16(false, a.h, false, b2, (short)0, acc2, false, false);
        acc3 = __builtin_amdgcn_wmma_f32_16x16x32_f16(false, a.h, false, b3, (short)0, acc3, false, false);

        __syncthreads();   // all waves done with this buffer before it is re-staged
    }

    // C layout: acc[r] for lane -> row m0 + half*8 + r, col n0 + j*16 + l16
    #pragma unroll
    for (int r = 0; r < 8; ++r) {
        float* crow = C + (size_t)(m0 + half * 8 + r) * ldC + n0 + l16;
        crow[0]  = acc0[r];
        crow[16] = acc1[r];
        crow[32] = acc2[r];
        crow[48] = acc3[r];
    }
}

// ---------------------------------------------------------------------------
// 5) causal depthwise conv (K=4) + SiLU over xz[:, :, 0:DI]
// ---------------------------------------------------------------------------
__global__ void k_conv_silu(const float* __restrict__ xz,
                            const float* __restrict__ conv_w,
                            const float* __restrict__ conv_b,
                            float* __restrict__ xc,
                            half_t* __restrict__ xc_h) {
    const size_t t = (size_t)blockIdx.x * blockDim.x + threadIdx.x;  // over ROWS*DI
    const int d = (int)(t % D_INNER);
    const int row = (int)(t / D_INNER);
    const int s = row % SEQ;
    const int b = row / SEQ;
    float acc = conv_b[d];
    #pragma unroll
    for (int k = 0; k < K_CONV; ++k) {
        const int sp = s + k - (K_CONV - 1);
        if (sp >= 0)
            acc = fmaf(xz[((size_t)(b * SEQ + sp)) * (2 * D_INNER) + d], conv_w[d * K_CONV + k], acc);
    }
    const float v = siluf_(acc);
    xc[t] = v;
    xc_h[t] = (half_t)v;
}

// dt slice of x_dbl -> f16 A operand for dt_proj
__global__ void k_dt_cvt(const float* __restrict__ xdbl, half_t* __restrict__ dt_h) {
    int i = blockIdx.x * blockDim.x + threadIdx.x;     // over ROWS*DT_RANK
    int row = i / DT_RANK, r = i % DT_RANK;
    dt_h[i] = (half_t)xdbl[(size_t)row * XP_PAD + r];
}

// delta = softplus(delta_pre + dt_proj_b)
__global__ void k_softplus(float* __restrict__ delta, const float* __restrict__ dt_b) {
    const size_t t = (size_t)blockIdx.x * blockDim.x + threadIdx.x;  // ROWS*DI
    const int d = (int)(t % D_INNER);
    delta[t] = softplusf_(delta[t] + dt_b[d]);
}

// ---------------------------------------------------------------------------
// 6) selective scan: one thread per (b,d) channel; h[N] in registers.
//    Fuses D-skip and silu(z) gate; emits f16 A-operand for out_proj.
// ---------------------------------------------------------------------------
__global__ void k_scan(const float* __restrict__ delta,
                       const float* __restrict__ xc,
                       const float* __restrict__ xz,
                       const float* __restrict__ xdbl,
                       const float* __restrict__ A_log,
                       const float* __restrict__ D_skip,
                       half_t* __restrict__ y_h) {
    const int chan = blockIdx.x * blockDim.x + threadIdx.x;  // 0 .. B*DI-1
    const int b = chan / D_INNER;
    const int d = chan % D_INNER;

    float Ar[N_STATE], h[N_STATE];
    #pragma unroll
    for (int n = 0; n < N_STATE; ++n) {
        Ar[n] = -__expf(A_log[d * N_STATE + n]);
        h[n] = 0.0f;
    }
    const float dsk = D_skip[d];

    for (int s = 0; s < SEQ; ++s) {
        const size_t row = (size_t)(b * SEQ + s);
        const float dlt = delta[row * D_INNER + d];
        const float u   = xc[row * D_INNER + d];
        const float zv  = xz[row * (2 * D_INNER) + D_INNER + d];
        const float du  = dlt * u;
        const float* bc = xdbl + row * XP_PAD + DT_RANK;   // B at +0..15, C at +16..31
        float y = 0.0f;
        #pragma unroll
        for (int n = 0; n < N_STATE; ++n) {
            h[n] = fmaf(__expf(dlt * Ar[n]), h[n], du * bc[n]);
            y = fmaf(h[n], bc[N_STATE + n], y);
        }
        const float out = (y + u * dsk) * siluf_(zv);
        y_h[row * D_INNER + d] = (half_t)out;
    }
}

// ---------------------------------------------------------------------------
// 7) residual + gate:  out = x + gate[b,:] * yproj
// ---------------------------------------------------------------------------
__global__ void k_final(const float* __restrict__ x,
                        const float* __restrict__ mod,
                        const float* __restrict__ yproj,
                        float* __restrict__ out) {
    const size_t t = (size_t)blockIdx.x * blockDim.x + threadIdx.x;  // ROWS*D
    const int d = (int)(t % D_MODEL);
    const int b = (int)(t / ((size_t)SEQ * D_MODEL));
    out[t] = fmaf(mod[b * 3 * D_MODEL + 2 * D_MODEL + d], yproj[t], x[t]);
}

// ---------------------------------------------------------------------------
extern "C" void kernel_launch(void* const* d_in, const int* in_sizes, int n_in,
                              void* d_out, int out_size, void* d_ws, size_t ws_size,
                              hipStream_t stream) {
    const float* x         = (const float*)d_in[0];
    const float* c         = (const float*)d_in[1];
    // d_in[2] ("w") is unused by the reference
    const float* ln_w      = (const float*)d_in[3];
    const float* ln_b      = (const float*)d_in[4];
    const float* ada_w     = (const float*)d_in[5];
    const float* ada_b     = (const float*)d_in[6];
    const float* in_proj_w = (const float*)d_in[7];
    const float* conv_w    = (const float*)d_in[8];
    const float* conv_b    = (const float*)d_in[9];
    const float* x_proj_w  = (const float*)d_in[10];
    const float* dt_proj_w = (const float*)d_in[11];
    const float* dt_proj_b = (const float*)d_in[12];
    const float* A_log     = (const float*)d_in[13];
    const float* D_skip    = (const float*)d_in[14];
    const float* out_proj_w= (const float*)d_in[15];
    float* out = (float*)d_out;

    // ---- workspace carve-up -------------------------------------------------
    char* base = (char*)d_ws;
    size_t off = 0;
    auto carve = [&](size_t bytes) -> char* {
        char* p = base + off;
        off = (off + bytes + 255) & ~(size_t)255;
        return p;
    };
    float*  mod     = (float*) carve((size_t)BATCH * 3 * D_MODEL * 4);
    half_t* xs_h    = (half_t*)carve((size_t)ROWS * D_MODEL * 2);
    half_t* w_in_h  = (half_t*)carve((size_t)2 * D_INNER * D_MODEL * 2);
    float*  xz      = (float*) carve((size_t)ROWS * 2 * D_INNER * 4);
    float*  xc      = (float*) carve((size_t)ROWS * D_INNER * 4);
    half_t* xc_h    = (half_t*)carve((size_t)ROWS * D_INNER * 2);
    half_t* w_xp_h  = (half_t*)carve((size_t)XP_PAD * D_INNER * 2);
    float*  xdbl    = (float*) carve((size_t)ROWS * XP_PAD * 4);
    half_t* dt_h    = (half_t*)carve((size_t)ROWS * DT_RANK * 2);
    half_t* w_dt_h  = (half_t*)carve((size_t)D_INNER * DT_RANK * 2);
    float*  delta   = (float*) carve((size_t)ROWS * D_INNER * 4);
    half_t* y_h     = (half_t*)carve((size_t)ROWS * D_INNER * 2);
    half_t* w_out_h = (half_t*)carve((size_t)D_MODEL * D_INNER * 2);
    float*  yproj   = (float*) carve((size_t)ROWS * D_MODEL * 4);

    // ---- 1) adaLN modulation ------------------------------------------------
    k_adaln<<<dim3(3 * D_MODEL / 256, BATCH), 256, 0, stream>>>(c, ada_w, ada_b, mod);

    // ---- 2) LN + modulate -> f16 -------------------------------------------
    k_ln_mod<<<ROWS, 256, 0, stream>>>(x, ln_w, ln_b, mod, xs_h);

    // ---- 3) weight repacks --------------------------------------------------
    {
        int n = 2 * D_INNER * D_MODEL;
        k_cvt16<<<(n + 255) / 256, 256, 0, stream>>>(in_proj_w, w_in_h, n);
        n = D_INNER * DT_RANK;
        k_cvt16<<<(n + 255) / 256, 256, 0, stream>>>(dt_proj_w, w_dt_h, n);
        n = D_MODEL * D_INNER;
        k_cvt16<<<(n + 255) / 256, 256, 0, stream>>>(out_proj_w, w_out_h, n);
        n = XP_PAD * D_INNER;
        k_cvt16_padrows<<<(n + 255) / 256, 256, 0, stream>>>(x_proj_w, w_xp_h);
    }

    // ---- 4) in_proj GEMM: [4096,1024] x [4096,1024]^T -> xz [4096,4096] ----
    {
        const int tilesN = (2 * D_INNER) / 64;          // 64
        const int blocks = (ROWS / 128) * tilesN;       // 2048
        k_wmma_gemm<<<blocks, 256, 0, stream>>>(xs_h, w_in_h, xz,
                                                D_MODEL, 2 * D_INNER, tilesN);
    }

    // ---- 5) conv + silu -----------------------------------------------------
    {
        size_t n = (size_t)ROWS * D_INNER;
        k_conv_silu<<<(unsigned)(n / 256), 256, 0, stream>>>(xz, conv_w, conv_b, xc, xc_h);
    }

    // ---- 6) x_proj GEMM: [4096,2048] x [128,2048]^T -> xdbl [4096,128] -----
    {
        const int tilesN = XP_PAD / 64;                 // 2
        const int blocks = (ROWS / 128) * tilesN;       // 64
        k_wmma_gemm<<<blocks, 256, 0, stream>>>(xc_h, w_xp_h, xdbl,
                                                D_INNER, XP_PAD, tilesN);
    }

    // ---- 7) dt slice -> f16, dt_proj GEMM, softplus ------------------------
    {
        int n = ROWS * DT_RANK;
        k_dt_cvt<<<(n + 255) / 256, 256, 0, stream>>>(xdbl, dt_h);
        const int tilesN = D_INNER / 64;                // 32
        const int blocks = (ROWS / 128) * tilesN;       // 1024
        k_wmma_gemm<<<blocks, 256, 0, stream>>>(dt_h, w_dt_h, delta,
                                                DT_RANK, D_INNER, tilesN);
        size_t ne = (size_t)ROWS * D_INNER;
        k_softplus<<<(unsigned)(ne / 256), 256, 0, stream>>>(delta, dt_proj_b);
    }

    // ---- 8) selective scan (register-resident state) -----------------------
    k_scan<<<(BATCH * D_INNER) / 256, 256, 0, stream>>>(delta, xc, xz, xdbl,
                                                        A_log, D_skip, y_h);

    // ---- 9) out_proj GEMM: [4096,2048] x [1024,2048]^T -> yproj [4096,1024]
    {
        const int tilesN = D_MODEL / 64;                // 16
        const int blocks = (ROWS / 128) * tilesN;       // 512
        k_wmma_gemm<<<blocks, 256, 0, stream>>>(y_h, w_out_h, yproj,
                                                D_INNER, D_MODEL, tilesN);
    }

    // ---- 10) residual + gate ------------------------------------------------
    {
        size_t n = (size_t)ROWS * D_MODEL;
        k_final<<<(unsigned)(n / 256), 256, 0, stream>>>(x, mod, yproj, out);
    }
}
